// CustomRNN_47923245089399
// MI455X (gfx1250) — matrix operand order repaired and checked by reading, not compiled, not run on hardware
//
#include <hip/hip_runtime.h>
#include <hip/hip_bf16.h>
#include <math.h>

// ---------------- types for WMMA ----------------
typedef __attribute__((ext_vector_type(16))) __bf16 v16bf;
typedef __attribute__((ext_vector_type(8)))  __bf16 v8bf;
typedef __attribute__((ext_vector_type(8)))  float  v8f;

union AFrag { v16bf v; v8bf h[2]; };

// ---------------- problem dims ----------------
#define BATCH   64
#define SEQ     512
#define INP     256
#define HID     1024
#define OUT     256
#define MROWS   (BATCH * SEQ)        // 32768 flat rows for the big GEMMs

// ---------------- workspace layout (bytes) ----------------
static constexpr size_t OFF_SYNC = 0;                                   // 2 u32 (pad 256)
static constexpr size_t OFF_XW   = 256;                                 // fp32 [B*S, H]
static constexpr size_t OFF_HS   = OFF_XW   + (size_t)MROWS*HID*4;      // bf16 [B*S, H]
static constexpr size_t OFF_WXHT = OFF_HS   + (size_t)MROWS*HID*2;      // bf16 [H, I] K-contig
static constexpr size_t OFF_WHHT = OFF_WXHT + (size_t)HID*INP*2;        // bf16 [H, H] K-contig
static constexpr size_t OFF_WHYT = OFF_WHHT + (size_t)HID*HID*2;        // bf16 [O, H] K-contig
static constexpr size_t OFF_H0   = OFF_WHYT + (size_t)OUT*HID*2;        // bf16 [B, H]
static constexpr size_t OFF_H1   = OFF_H0   + (size_t)BATCH*HID*2;

// =================================================================
// helpers
// =================================================================
__device__ __forceinline__ __bf16 to_bf16(float f) { return (__bf16)f; }

// LDS byte offset of a generic pointer to __shared__ memory
// (async-to-LDS instructions take dsaddr relative to the wave's LDS base)
__device__ __forceinline__ unsigned lds_offset(const void* p) {
  return (unsigned)(size_t)(__attribute__((address_space(3))) const char*)(const char*)p;
}

// per-lane async copy of 16 bytes global -> LDS, tracked with ASYNCcnt
__device__ __forceinline__ void async_copy_b128(unsigned lds_off, const void* gaddr) {
  asm volatile("global_load_async_to_lds_b128 %0, %1, off"
               :: "v"(lds_off), "v"(gaddr)
               : "memory");
}

__device__ __forceinline__ void wait_asynccnt0() {
  asm volatile("s_wait_asynccnt 0" ::: "memory");
}

__global__ void init_sync_kernel(unsigned* s) { s[0] = 0u; s[1] = 0u; }

// dst[c*R + r] = bf16(src[r*C + c])  (make contraction dim contiguous per column)
__global__ void transpose_bf16_kernel(const float* __restrict__ src,
                                      __bf16* __restrict__ dst, int R, int C) {
  int id = blockIdx.x * blockDim.x + threadIdx.x;
  if (id < R * C) {
    int r = id % R;
    int c = id / R;
    dst[id] = to_bf16(src[(size_t)r * C + c]);
  }
}

__global__ void f32_to_bf16_kernel(const float* __restrict__ s,
                                   __bf16* __restrict__ d, int n) {
  int id = blockIdx.x * blockDim.x + threadIdx.x;
  if (id < n) d[id] = to_bf16(s[id]);
}

// device-wide generation barrier (release/acquire)
__device__ __forceinline__ void grid_sync(unsigned* cnt, unsigned* gen, unsigned nb) {
  __threadfence();                 // release: flush this thread's global stores
  __syncthreads();
  if (threadIdx.x == 0) {
    unsigned g = __hip_atomic_load(gen, __ATOMIC_RELAXED, __HIP_MEMORY_SCOPE_AGENT);
    if (__hip_atomic_fetch_add(cnt, 1u, __ATOMIC_ACQ_REL, __HIP_MEMORY_SCOPE_AGENT) == nb - 1u) {
      __hip_atomic_store(cnt, 0u, __ATOMIC_RELAXED, __HIP_MEMORY_SCOPE_AGENT);
      __hip_atomic_fetch_add(gen, 1u, __ATOMIC_RELEASE, __HIP_MEMORY_SCOPE_AGENT);
    } else {
      while (__hip_atomic_load(gen, __ATOMIC_ACQUIRE, __HIP_MEMORY_SCOPE_AGENT) == g) {
        __builtin_amdgcn_s_sleep(8);
      }
    }
  }
  __syncthreads();
  __threadfence();                 // acquire side
}

// =================================================================
// Kernel A: xW[m, n] = x[m, :I] @ Wxh[:, n] + bh[n]   (m = b*S+s)
// one 16x16 tile per wave, K = 256 => 8 WMMAs (bf16 in, f32 acc)
// =================================================================
__global__ void __launch_bounds__(256) xw_gemm_kernel(
    const float* __restrict__ x, const __bf16* __restrict__ WxhT,
    const float* __restrict__ bh, float* __restrict__ xW) {
  int wave  = (blockIdx.x * blockDim.x + threadIdx.x) >> 5;
  int lane  = threadIdx.x & 31;
  int Ntile = wave & 63;
  int Mtile = wave >> 6;
  int m0 = Mtile * 16, n0 = Ntile * 16;
  int half = lane >> 4, nl = lane & 15;

  float bhv = bh[n0 + nl];
  v8f c;
#pragma unroll
  for (int r = 0; r < 8; ++r) c[r] = bhv;

  const float*  xrow = x    + (size_t)(m0 + nl) * INP;
  const __bf16* bcol = WxhT + (size_t)(n0 + nl) * INP;

#pragma unroll
  for (int kc = 0; kc < 8; ++kc) {
    int kb = kc * 32 + half * 16;
    float tmp[16];
    *(float4*)(tmp + 0)  = *(const float4*)(xrow + kb);
    *(float4*)(tmp + 4)  = *(const float4*)(xrow + kb + 4);
    *(float4*)(tmp + 8)  = *(const float4*)(xrow + kb + 8);
    *(float4*)(tmp + 12) = *(const float4*)(xrow + kb + 12);
    v16bf a;
#pragma unroll
    for (int i = 0; i < 16; ++i) a[i] = to_bf16(tmp[i]);
    v16bf b = *(const v16bf*)(bcol + kc * 32 + half * 16);
    c = __builtin_amdgcn_wmma_f32_16x16x32_bf16(false, a, false, b,
                                                (short)0, c, false, false);
  }
#pragma unroll
  for (int r = 0; r < 8; ++r)
    xW[(size_t)(m0 + half * 8 + r) * HID + n0 + nl] = c[r];
}

// =================================================================
// Persistent recurrence kernel: 512 steps of h = tanh(xW_t + h @ Whh)
// 32 blocks x 256 threads; block b owns N columns [32b, 32b+32) with
// the Whh^T slice (64 KB bf16) staged to LDS via async-to-LDS DMA.
// wave w: Mtile = w&3, local Ntile = w>>2  (4 M-tiles x 2 N-tiles)
// =================================================================
__global__ void __launch_bounds__(256) rnn_steps_kernel(
    const float* __restrict__ xW, const __bf16* __restrict__ WhhT,
    __bf16* __restrict__ h0, __bf16* __restrict__ h1,
    __bf16* __restrict__ hs, float* __restrict__ hfinal,
    unsigned* __restrict__ sync) {
  extern __shared__ __bf16 sB[];                 // [32 cols][1024 K] bf16 = 64 KB
  const int b     = blockIdx.x;                  // 0..31
  const int nbase = b * 32;

  // stage Whh^T slice into LDS with global_load_async_to_lds_b128 (ASYNCcnt)
  {
    const char* gsrc = (const char*)(WhhT + (size_t)nbase * HID);
    char*       lbase = (char*)sB;
#pragma unroll
    for (int i = 0; i < 16; ++i) {                       // 16 x 256 threads x 16B = 64 KB
      size_t off = ((size_t)i * 256 + threadIdx.x) * 16;
      async_copy_b128(lds_offset(lbase + off), gsrc + off);
    }
    wait_asynccnt0();
  }
  __syncthreads();

  const int wave = threadIdx.x >> 5, lane = threadIdx.x & 31;
  const int m0 = (wave & 3) * 16;                // batch-row tile
  const int nt = wave >> 2;                      // local N tile (0 or 1)
  const int n0 = nbase + nt * 16;
  const int half = lane >> 4, nl = lane & 15;

  const __bf16* sbcol = sB + (size_t)(nt * 16 + nl) * HID;
  unsigned* cnt = sync;
  unsigned* gen = sync + 1;

#pragma unroll 1
  for (int t = 0; t < SEQ; ++t) {
    const __bf16* hc = (t & 1) ? h1 : h0;
    __bf16*       hn = (t & 1) ? h0 : h1;

    // C tile <- pre-activation xW[batch, t, n]
    v8f c;
    const float* xwp = xW + (size_t)t * HID + n0 + nl;
#pragma unroll
    for (int r = 0; r < 8; ++r)
      c[r] = xwp[(size_t)(m0 + half * 8 + r) * ((size_t)SEQ * HID)];

    // prefetch next timestep's pre-activation tile (overlaps the WMMA chain)
    if (t + 1 < SEQ) {
#pragma unroll
      for (int r = 0; r < 8; ++r)
        __builtin_prefetch(xwp + HID + (size_t)(m0 + half * 8 + r) * ((size_t)SEQ * HID), 0, 0);
    }

    const __bf16* arow = hc + (size_t)(m0 + nl) * HID;
#pragma unroll 8
    for (int kc = 0; kc < 32; ++kc) {
      int kb = kc * 32 + half * 16;
      AFrag af;
      af.h[0] = *(const v8bf*)(arow + kb);
      af.h[1] = *(const v8bf*)(arow + kb + 8);
      v16bf bf = *(const v16bf*)(sbcol + kc * 32 + half * 16);
      c = __builtin_amdgcn_wmma_f32_16x16x32_bf16(false, af.v, false, bf,
                                                  (short)0, c, false, false);
    }

#pragma unroll
    for (int r = 0; r < 8; ++r) {
      float f = tanhf(c[r]);
      int mrow = m0 + half * 8 + r;              // batch index
      __bf16 hb = to_bf16(f);
      hn[(size_t)mrow * HID + n0 + nl] = hb;
      hs[(size_t)mrow * ((size_t)SEQ * HID) + (size_t)t * HID + n0 + nl] = hb;
      if (t == SEQ - 1) hfinal[(size_t)mrow * HID + n0 + nl] = f;
    }

    grid_sync(cnt, gen, 32u);
  }
}

// =================================================================
// Kernel C: out[m, o] = hs[m, :H] @ Why[:, o] + by[o]
// K = 1024 => 32 WMMAs; N = 256 => 16 Ntiles; 32768 waves = 4096 blocks
// =================================================================
__global__ void __launch_bounds__(256) out_gemm_kernel(
    const __bf16* __restrict__ hs, const __bf16* __restrict__ WhyT,
    const float* __restrict__ by, float* __restrict__ out) {
  int wave  = (blockIdx.x * blockDim.x + threadIdx.x) >> 5;
  int lane  = threadIdx.x & 31;
  int Ntile = wave & 15;
  int Mtile = wave >> 4;
  int m0 = Mtile * 16, n0 = Ntile * 16;
  int half = lane >> 4, nl = lane & 15;

  float byv = by[n0 + nl];
  v8f c;
#pragma unroll
  for (int r = 0; r < 8; ++r) c[r] = byv;

  const __bf16* arow = hs   + (size_t)(m0 + nl) * HID;
  const __bf16* bcol = WhyT + (size_t)(n0 + nl) * HID;

#pragma unroll 8
  for (int kc = 0; kc < 32; ++kc) {
    int kb = kc * 32 + half * 16;
    AFrag af;
    af.h[0] = *(const v8bf*)(arow + kb);
    af.h[1] = *(const v8bf*)(arow + kb + 8);
    v16bf b = *(const v16bf*)(bcol + kc * 32 + half * 16);
    c = __builtin_amdgcn_wmma_f32_16x16x32_bf16(false, af.v, false, b,
                                                (short)0, c, false, false);
  }
#pragma unroll
  for (int r = 0; r < 8; ++r)
    out[(size_t)(m0 + half * 8 + r) * OUT + n0 + nl] = c[r];
}

// =================================================================
extern "C" void kernel_launch(void* const* d_in, const int* in_sizes, int n_in,
                              void* d_out, int out_size, void* d_ws, size_t ws_size,
                              hipStream_t stream) {
  (void)in_sizes; (void)n_in; (void)out_size; (void)ws_size;

  const float* x      = (const float*)d_in[0];
  const float* h_prev = (const float*)d_in[1];
  const float* Wxh    = (const float*)d_in[2];
  const float* Whh    = (const float*)d_in[3];
  const float* Why    = (const float*)d_in[4];
  const float* bh     = (const float*)d_in[5];
  const float* by     = (const float*)d_in[6];

  char* ws = (char*)d_ws;
  unsigned* sync   = (unsigned*)(ws + OFF_SYNC);
  float*    xWbuf  = (float*)   (ws + OFF_XW);
  __bf16*   hsbuf  = (__bf16*)  (ws + OFF_HS);
  __bf16*   WxhT   = (__bf16*)  (ws + OFF_WXHT);
  __bf16*   WhhT   = (__bf16*)  (ws + OFF_WHHT);
  __bf16*   WhyT   = (__bf16*)  (ws + OFF_WHYT);
  __bf16*   h0     = (__bf16*)  (ws + OFF_H0);
  __bf16*   h1     = (__bf16*)  (ws + OFF_H1);

  float* out_y = (float*)d_out;                              // (B,S,O) fp32
  float* out_h = (float*)d_out + (size_t)BATCH * SEQ * OUT;  // (B,H) fp32

  init_sync_kernel<<<1, 1, 0, stream>>>(sync);

  transpose_bf16_kernel<<<(INP * HID + 255) / 256, 256, 0, stream>>>(Wxh, WxhT, INP, HID);
  transpose_bf16_kernel<<<(HID * HID + 255) / 256, 256, 0, stream>>>(Whh, WhhT, HID, HID);
  transpose_bf16_kernel<<<(HID * OUT + 255) / 256, 256, 0, stream>>>(Why, WhyT, HID, OUT);
  f32_to_bf16_kernel<<<(BATCH * HID + 255) / 256, 256, 0, stream>>>(h_prev, h0, BATCH * HID);

  // xW = x @ Wxh + bh : 131072 tiles -> 16384 blocks of 8 waves
  xw_gemm_kernel<<<16384, 256, 0, stream>>>(x, WxhT, bh, xWbuf);

  // persistent scan: 32 blocks, 64 KB dynamic LDS each (Whh^T slice)
  rnn_steps_kernel<<<32, 256, 32 * HID * 2, stream>>>(xWbuf, WhhT, h0, h1,
                                                      hsbuf, out_h, sync);

  // outputs = hs @ Why + by : 32768 tiles -> 4096 blocks of 8 waves
  out_gemm_kernel<<<4096, 256, 0, stream>>>(hsbuf, WhyT, by, out_y);
}
